// PointerGenerator_88991722373682
// MI455X (gfx1250) — matrix operand order repaired and checked by reading, not compiled, not run on hardware
//
#include <hip/hip_runtime.h>
#include <hip/hip_bf16.h>
#include <math.h>

#define VOCAB 32000
#define Bsz   4
#define Tdim  512
#define Sdim  512
#define Hdim  1024
#define Ddim  513
#define ROWS  (Bsz * Tdim)          // 2048
#define CCOLS (VOCAB + 2 * Ddim)    // 33026
#define EXCL  (VOCAB + Ddim)        // 32513 (zeroed for argmax only)
#define KSTEP 64                    // K per LDS stage

typedef __attribute__((ext_vector_type(16))) __bf16 v16bf;
typedef __attribute__((ext_vector_type(8)))  __bf16 v8bf;
typedef __attribute__((ext_vector_type(8)))  float  v8f;
typedef __attribute__((ext_vector_type(2)))  float  v2f;

// ---------------------------------------------------------------- CDNA5 async copy (ASYNCcnt)
__device__ __forceinline__ void async_copy_b128(unsigned lds_addr, const void* gaddr) {
  asm volatile("global_load_async_to_lds_b128 %0, %1, off"
               :: "v"(lds_addr), "v"((unsigned long long)(uintptr_t)gaddr)
               : "memory");
}
__device__ __forceinline__ void wait_asynccnt0() {
  asm volatile("s_wait_asynccnt 0x0" ::: "memory");
}

// ---------------------------------------------------------------- converts
__global__ void k_f32_to_bf16(const float* __restrict__ in,
                              unsigned short* __restrict__ out, int n) {
  int i = blockIdx.x * blockDim.x + threadIdx.x;
  if (i < n) {
    unsigned u = __float_as_uint(in[i]);
    unsigned r = u + 0x7FFFu + ((u >> 16) & 1u);   // RNE to bf16
    out[i] = (unsigned short)(r >> 16);
  }
}

// ---------------------------------------------------------------- switch softmax
__global__ void k_switch(const float* __restrict__ hid, const float* __restrict__ Wsw,
                         const float* __restrict__ bsw, float* __restrict__ p) {
  __shared__ float red[3][128];
  int row = blockIdx.x, tid = threadIdx.x;
  const float* h = hid + (size_t)row * Hdim;
  float s0 = 0.f, s1 = 0.f, s2 = 0.f;
  for (int k = tid; k < Hdim; k += 128) {
    float x = h[k];
    s0 += x * Wsw[k]; s1 += x * Wsw[Hdim + k]; s2 += x * Wsw[2 * Hdim + k];
  }
  red[0][tid] = s0; red[1][tid] = s1; red[2][tid] = s2;
  __syncthreads();
  for (int st = 64; st > 0; st >>= 1) {
    if (tid < st) {
      red[0][tid] += red[0][tid + st];
      red[1][tid] += red[1][tid + st];
      red[2][tid] += red[2][tid + st];
    }
    __syncthreads();
  }
  if (tid == 0) {
    float l0 = red[0][0] + bsw[0], l1 = red[1][0] + bsw[1], l2 = red[2][0] + bsw[2];
    float m = fmaxf(l0, fmaxf(l1, l2));
    float e0 = __expf(l0 - m), e1 = __expf(l1 - m), e2 = __expf(l2 - m);
    float inv = 1.f / (e0 + e1 + e2);
    p[row * 3 + 0] = e0 * inv; p[row * 3 + 1] = e1 * inv; p[row * 3 + 2] = e2 * inv;
  }
}

// ---------------------------------------------------------------- LDS fragment loads
// A fragment (16x32 bf16): lane = half*16+m ; elems 0-7 = K[half*8..], 8-15 = K[16+half*8..]
__device__ __forceinline__ v16bf lds_a_frag(const __bf16* base, int row, int kk, int half) {
  const __bf16* rp = base + row * KSTEP + kk;
  v8bf lo = *reinterpret_cast<const v8bf*>(rp + half * 8);
  v8bf hi = *reinterpret_cast<const v8bf*>(rp + 16 + half * 8);
  return __builtin_shufflevector(lo, hi, 0,1,2,3,4,5,6,7,8,9,10,11,12,13,14,15);
}
// B fragment (32x16 bf16, B[k][n] = W[n][k]): lane n, elems 0-7 = K[half*8..], 8-15 = K[16+half*8..]
__device__ __forceinline__ v16bf lds_b_frag(const __bf16* base, int row, int kk, int half) {
  const __bf16* rp = base + row * KSTEP + kk;
  v8bf lo = *reinterpret_cast<const v8bf*>(rp + half * 8);
  v8bf hi = *reinterpret_cast<const v8bf*>(rp + 16 + half * 8);
  return __builtin_shufflevector(lo, hi, 0,1,2,3,4,5,6,7,8,9,10,11,12,13,14,15);
}

// ---------------------------------------------------------------- bf16 WMMA GEMM w/ async LDS pipeline
// block = 128 threads = 4 waves; block tile 64x64; K staged 64 at a time, double-buffered.
__global__ void __launch_bounds__(128)
k_gemm_vocab(const __bf16* __restrict__ hidb, const __bf16* __restrict__ wvb,
             const float* __restrict__ bvocab, float* __restrict__ out) {
  __shared__ __bf16 Abuf[2][64 * KSTEP];   // 8KB per buffer
  __shared__ __bf16 Bbuf[2][64 * KSTEP];   // 8KB per buffer
  int tid = threadIdx.x;
  int wave = tid >> 5, lane = tid & 31;
  int half = lane >> 4, n16 = lane & 15;
  int rowblk = blockIdx.y * 64, colblk = blockIdx.x * 64;
  const __bf16* ga = hidb + (size_t)rowblk * Hdim;   // A tile rows
  const __bf16* gb = wvb  + (size_t)colblk * Hdim;   // B tile rows (W_vocab row-major)

  // tile = 64 rows x 128B (KSTEP bf16) = 512 chunks of 16B; each thread copies 4 A + 4 B chunks
  unsigned abase[2] = { (unsigned)(uintptr_t)&Abuf[0][0], (unsigned)(uintptr_t)&Abuf[1][0] };
  unsigned bbase[2] = { (unsigned)(uintptr_t)&Bbuf[0][0], (unsigned)(uintptr_t)&Bbuf[1][0] };

  auto issue_stage = [&](int buf, int kb) {
#pragma unroll
    for (int i = 0; i < 4; i++) {
      int c = tid + i * 128;
      int r = c >> 3, k8 = (c & 7) * 8;
      async_copy_b128(abase[buf] + (unsigned)c * 16u, ga + (size_t)r * Hdim + kb + k8);
      async_copy_b128(bbase[buf] + (unsigned)c * 16u, gb + (size_t)r * Hdim + kb + k8);
    }
  };

  v8f acc[4] = {};
  const int nstages = Hdim / KSTEP;        // 16
  issue_stage(0, 0);
  for (int s = 0; s < nstages; ++s) {
    int cur = s & 1;
    wait_asynccnt0();                      // own stage-s copies landed
    __syncthreads();                       // everyone's copies landed, prev reads done
    if (s + 1 < nstages) issue_stage(cur ^ 1, (s + 1) * KSTEP);  // DMA next stage during compute
    const __bf16* Ab = &Abuf[cur][0];
    const __bf16* Bb = &Bbuf[cur][0];
#pragma unroll
    for (int kk = 0; kk < KSTEP; kk += 32) {
      v16bf af = lds_a_frag(Ab, wave * 16 + n16, kk, half);
#pragma unroll
      for (int j = 0; j < 4; j++) {
        v16bf bfrag = lds_b_frag(Bb, j * 16 + n16, kk, half);
        acc[j] = __builtin_amdgcn_wmma_f32_16x16x32_bf16(
            false, af, false, bfrag, (short)0, acc[j], false, false);
      }
    }
    __syncthreads();                       // done reading buf[cur] before it is overwritten
  }

  const float NEG_INF = -__builtin_inff();
  int rowbase = rowblk + wave * 16;
#pragma unroll
  for (int j = 0; j < 4; j++) {
    int col = colblk + j * 16 + n16;
    float bias = bvocab[col];
#pragma unroll
    for (int v = 0; v < 8; v++) {
      int r = rowbase + v + 8 * half;
      float val = acc[j][v] + bias;
      out[(size_t)r * CCOLS + col] = (col == 0) ? NEG_INF : val;
    }
  }
}

// ---------------------------------------------------------------- vocab softmax * p_gen
__global__ void k_vocab_softmax(float* __restrict__ out, const float* __restrict__ p) {
  __shared__ float red[256];
  int row = blockIdx.x, tid = threadIdx.x;
  float* rp = out + (size_t)row * CCOLS;
  float mx = -__builtin_inff();
  for (int c = tid; c < VOCAB; c += 256) mx = fmaxf(mx, rp[c]);
  red[tid] = mx; __syncthreads();
  for (int st = 128; st > 0; st >>= 1) {
    if (tid < st) red[tid] = fmaxf(red[tid], red[tid + st]);
    __syncthreads();
  }
  mx = red[0]; __syncthreads();
  float s = 0.f;
  for (int c = tid; c < VOCAB; c += 256) s += __expf(rp[c] - mx);
  red[tid] = s; __syncthreads();
  for (int st = 128; st > 0; st >>= 1) {
    if (tid < st) red[tid] += red[tid + st];
    __syncthreads();
  }
  float scale = p[row * 3 + 2] / red[0];
  for (int c = tid; c < VOCAB; c += 256) rp[c] = __expf(rp[c] - mx) * scale;
}

// ---------------------------------------------------------------- copy-attn (exact f32 WMMA)
// out[b,t,coloff+d] = sum_s att[b,t,s]*p[b,t] * maps[b,s,d]; one 16x16 tile per wave
__global__ void k_copy_attn(const float* __restrict__ att, const int* __restrict__ maps,
                            const float* __restrict__ p, int pidx, int coloff,
                            float* __restrict__ out) {
  int wave = threadIdx.x >> 5, lane = threadIdx.x & 31;
  int half = lane >> 4, n16 = lane & 15;
  int tile = blockIdx.x * 4 + wave;               // 0..4223
  int b = tile / (32 * 33);
  int rem = tile % (32 * 33);
  int rowbase = (rem / 33) * 16;
  int colbase = (rem % 33) * 16;
  int col = colbase + n16;
  float msk = (col < Ddim) ? 1.f : 0.f;
  int colc = (col < Ddim) ? col : 0;              // clamped, branchless (EXEC stays all-1)
  const float* arow = att + (size_t)(b * Tdim + rowbase + n16) * Sdim;
  float pv = p[(b * Tdim + rowbase + n16) * 3 + pidx];
  const int* mcol = maps + (size_t)b * Sdim * Ddim + colc;
  v8f acc = {};
  for (int k = 0; k < Sdim; k += 4) {
    v2f a = *reinterpret_cast<const v2f*>(arow + k + 2 * half);
    a = a * pv;
    v2f bb;
    bb.x = msk * (float)mcol[(size_t)(k + 2 * half) * Ddim];
    bb.y = msk * (float)mcol[(size_t)(k + 2 * half + 1) * Ddim];
    acc = __builtin_amdgcn_wmma_f32_16x16x4_f32(
        false, a, false, bb, (short)0, acc, false, false);
  }
  if (col < Ddim) {
#pragma unroll
    for (int v = 0; v < 8; v++) {
      int r = rowbase + v + 8 * half;
      out[(size_t)(b * Tdim + r) * CCOLS + coloff + col] = acc[v];
    }
  }
}

// ---------------------------------------------------------------- argmax (skip EXCL)
__global__ void k_argmax(const float* __restrict__ out, float* __restrict__ pred) {
  __shared__ float rv[256];
  __shared__ int   ri[256];
  int row = blockIdx.x, tid = threadIdx.x;
  const float* rp = out + (size_t)row * CCOLS;
  float best = -__builtin_inff(); int bidx = 0x7fffffff;
  for (int c = tid; c < CCOLS; c += 256) {
    if (c == EXCL) continue;
    float v = rp[c];
    if (v > best) { best = v; bidx = c; }         // strict > keeps first occurrence
  }
  rv[tid] = best; ri[tid] = bidx; __syncthreads();
  for (int st = 128; st > 0; st >>= 1) {
    if (tid < st) {
      float v2 = rv[tid + st]; int i2 = ri[tid + st];
      if (v2 > rv[tid] || (v2 == rv[tid] && i2 < ri[tid])) { rv[tid] = v2; ri[tid] = i2; }
    }
    __syncthreads();
  }
  if (tid == 0) pred[row] = (float)ri[0];
}

// ---------------------------------------------------------------- launch
extern "C" void kernel_launch(void* const* d_in, const int* in_sizes, int n_in,
                              void* d_out, int out_size, void* d_ws, size_t ws_size,
                              hipStream_t stream) {
  const float* hiddens  = (const float*)d_in[0];
  const float* src_att  = (const float*)d_in[1];
  const int*   src_map  = (const int*)  d_in[2];
  const float* tgt_att  = (const float*)d_in[3];
  const int*   tgt_map  = (const int*)  d_in[4];
  const float* W_vocab  = (const float*)d_in[5];
  const float* b_vocab  = (const float*)d_in[6];
  const float* W_switch = (const float*)d_in[7];
  const float* b_switch = (const float*)d_in[8];
  (void)in_sizes; (void)n_in; (void)out_size; (void)ws_size;

  float* out = (float*)d_out;
  char*  ws  = (char*)d_ws;
  float*          p_sw = (float*)ws;                                   // 2048*3 f32
  unsigned short* hidb = (unsigned short*)(ws + 32768);                // 2048*1024 bf16
  unsigned short* wvb  = (unsigned short*)(ws + 32768 + (size_t)ROWS * Hdim * 2);

  int nh = ROWS * Hdim;
  int nw = VOCAB * Hdim;
  k_f32_to_bf16<<<(nh + 255) / 256, 256, 0, stream>>>(hiddens, hidb, nh);
  k_f32_to_bf16<<<(nw + 255) / 256, 256, 0, stream>>>(W_vocab, wvb, nw);
  k_switch<<<ROWS, 128, 0, stream>>>(hiddens, W_switch, b_switch, p_sw);
  k_gemm_vocab<<<dim3(VOCAB / 64, ROWS / 64), 128, 0, stream>>>(
      (const __bf16*)hidb, (const __bf16*)wvb, b_vocab, out);
  k_vocab_softmax<<<ROWS, 256, 0, stream>>>(out, p_sw);
  k_copy_attn<<<1056, 128, 0, stream>>>(src_att, src_map, p_sw, 0, VOCAB, out);
  k_copy_attn<<<1056, 128, 0, stream>>>(tgt_att, tgt_map, p_sw, 1, VOCAB + Ddim, out);
  k_argmax<<<ROWS, 256, 0, stream>>>(out, out + (size_t)ROWS * CCOLS);
}
